// TriangleAttention_20298015441382
// MI455X (gfx1250) — compile-verified, matrix operand order checked by reading, not run on hardware
//
#include <hip/hip_runtime.h>
#include <hip/hip_bf16.h>
#include <cstdint>

// ---------- CDNA5 WMMA helpers (wave32, 16x16x32 f16 -> f32) ----------
typedef __attribute__((ext_vector_type(16))) _Float16 v16h;
typedef __attribute__((ext_vector_type(8)))  _Float16 v8h;
typedef __attribute__((ext_vector_type(8)))  float    v8f;
typedef unsigned int v4u __attribute__((ext_vector_type(4)));
typedef int          v8i __attribute__((ext_vector_type(8)));
typedef int          v4i __attribute__((ext_vector_type(4)));

// A/B operand packing for 16-bit WMMA (ISA 7.12.2):
// lane L holds row/col (L%16); halves e: K = (L/16)*8 + (e%8) + (e/8)*16.
// Each lane needs two contiguous 8-half (16B) runs at +0 and +16 halves.
__device__ __forceinline__ v16h load_ab(const _Float16* p) {
    v8h lo = *(const v8h*)(p);
    v8h hi = *(const v8h*)(p + 16);
    return __builtin_shufflevector(lo, hi, 0,1,2,3,4,5,6,7,8,9,10,11,12,13,14,15);
}

__device__ __forceinline__ v8f wmma16(v16h a, v16h b, v8f c) {
    return __builtin_amdgcn_wmma_f32_16x16x32_f16(
        false, a, false, b, (short)0, c, false, false);
}

// ---------- Tensor Data Mover: 2-D tile Global -> LDS (ISA ch.8 D# layout) ----
// dim0/stride0 in elements (f16, data_size code 1 = 2 bytes); dim1 rows.
// This toolchain exposes the 6-arg form: (v4u, v8i, v4i, v4i, v8i, i32 cpol).
__device__ __forceinline__ void tdm_load_2d(unsigned lds_off, const _Float16* gptr,
                                            unsigned dim0, unsigned dim1,
                                            unsigned long long stride0,
                                            unsigned tile0, unsigned tile1) {
    unsigned long long ga = (unsigned long long)(uintptr_t)gptr;
    v4u g0;
    g0[0] = 1u;                                        // count=1, user desc
    g0[1] = lds_off;                                   // lds_addr (bytes)
    g0[2] = (unsigned)(ga & 0xffffffffu);              // global_addr[31:0]
    g0[3] = (unsigned)((ga >> 32) & 0x01ffffffu) | (2u << 30);  // [56:32] | type=2
    v8i g1;
    g1[0] = (int)(1u << 16);                           // data_size=1 (2B), mask=0
    g1[1] = (int)((dim0 & 0xffffu) << 16);             // tensor_dim0[15:0]
    g1[2] = (int)(((dim0 >> 16) & 0xffffu) | ((dim1 & 0xffffu) << 16));
    g1[3] = (int)(((dim1 >> 16) & 0xffffu) | ((tile0 & 0xffffu) << 16));
    g1[4] = (int)(tile1 & 0xffffu);                    // tile_dim1, tile_dim2=0
    g1[5] = (int)(stride0 & 0xffffffffu);              // tensor_dim0_stride[31:0]
    g1[6] = (int)((stride0 >> 32) & 0xffffu);          // stride0[47:32], stride1=0
    g1[7] = 0;
    v4i g2 = {};
    v4i g3 = {};
    v8i g4 = {};
    __builtin_amdgcn_tensor_load_to_lds(g0, g1, g2, g3, g4, 0);
}

// Problem constants
#define NROW 65536   // B*I*J
#define CDIM 256
#define HDIM 512     // H*D
#define II   256
#define JJ   256
#define HH   8
#define DD   64

// ---------- Kernel T: transpose fp32 weight [R,Ccols] -> f16 [Ccols,R] ----------
__global__ void wtrans_kernel(const float* __restrict__ src, _Float16* __restrict__ dst,
                              int R, int Ccols) {
    size_t idx = (size_t)blockIdx.x * 256 + threadIdx.x;
    if (idx >= (size_t)R * Ccols) return;
    int r = (int)(idx / Ccols);
    int c = (int)(idx % Ccols);
    dst[(size_t)c * R + r] = (_Float16)src[idx];
}

// ---------- Kernel LN: LayerNorm over C, write f16 ----------
__global__ __launch_bounds__(256)
void ln_kernel(const float* __restrict__ x, const float* __restrict__ gamma,
               const float* __restrict__ beta, _Float16* __restrict__ xn) {
    __shared__ float red[256];
    size_t row = blockIdx.x;
    int c = threadIdx.x;
    float v = x[row * CDIM + c];
    red[c] = v;
    __syncthreads();
    for (int s = 128; s > 0; s >>= 1) {
        if (c < s) red[c] += red[c + s];
        __syncthreads();
    }
    float mu = red[0] * (1.0f / CDIM);
    __syncthreads();
    float d = v - mu;
    red[c] = d * d;
    __syncthreads();
    for (int s = 128; s > 0; s >>= 1) {
        if (c < s) red[c] += red[c + s];
        __syncthreads();
    }
    float var = red[0] * (1.0f / CDIM);
    float out = d * rsqrtf(var + 1e-5f) * gamma[c] + beta[c];
    xn[row * CDIM + c] = (_Float16)out;
}

// ---------- Kernel P: projection GEMM, wave computes 32x32 (2x2 WMMA tiles) ----
// mode 0: q (scale 1/8)  1: k  2: v (store transposed vt[(i*H+h)*64+d][j])  3: gate
__global__ __launch_bounds__(256)
void proj_kernel(const _Float16* __restrict__ xn, const _Float16* __restrict__ wt,
                 const float* __restrict__ bg, _Float16* __restrict__ out,
                 _Float16* __restrict__ vt, int mode) {
    int lane = threadIdx.x & 31;
    int wave = threadIdx.x >> 5;
    int tile = blockIdx.x * 8 + wave;       // 32768 wave tiles
    int tm = tile >> 4;                     // 2048 row tiles (32 rows)
    int tn = tile & 15;                     // 16 col tiles (32 cols)
    int row0 = tm * 32, col0 = tn * 32;
    int mrow = lane & 15;
    int kh   = (lane >> 4) * 8;
    int mhi  = (lane >> 4) << 3;

    const _Float16* a0p = xn + (size_t)(row0 + mrow) * CDIM + kh;
    const _Float16* a1p = a0p + (size_t)16 * CDIM;
    const _Float16* b0p = wt + (size_t)(col0 + mrow) * CDIM + kh;
    const _Float16* b1p = b0p + (size_t)16 * CDIM;

    v8f acc[2][2];
    acc[0][0] = {}; acc[0][1] = {}; acc[1][0] = {}; acc[1][1] = {};
#pragma unroll
    for (int kb = 0; kb < CDIM; kb += 32) {
        v16h a0 = load_ab(a0p + kb);
        v16h a1 = load_ab(a1p + kb);
        v16h b0 = load_ab(b0p + kb);
        v16h b1 = load_ab(b1p + kb);
        acc[0][0] = wmma16(a0, b0, acc[0][0]);
        acc[0][1] = wmma16(a0, b1, acc[0][1]);
        acc[1][0] = wmma16(a1, b0, acc[1][0]);
        acc[1][1] = wmma16(a1, b1, acc[1][1]);
    }

#pragma unroll
    for (int p = 0; p < 2; p++) {
#pragma unroll
        for (int qq = 0; qq < 2; qq++) {
            int colN = col0 + qq * 16 + mrow;
#pragma unroll
            for (int r = 0; r < 8; r++) {
                int row = row0 + p * 16 + r + mhi;
                float val = acc[p][qq][r];
                if (mode == 0) {
                    out[(size_t)row * HDIM + colN] = (_Float16)(val * 0.125f);
                } else if (mode == 1) {
                    out[(size_t)row * HDIM + colN] = (_Float16)val;
                } else if (mode == 2) {
                    int h = colN >> 6, d = colN & 63;
                    int i = row >> 8, j = row & 255;
                    vt[(((size_t)(i * HH + h) * DD + d) << 8) + j] = (_Float16)val;
                } else {
                    float s = 1.0f / (1.0f + __expf(-(val + bg[colN])));
                    out[(size_t)row * HDIM + colN] = (_Float16)s;
                }
            }
        }
    }
}

// ---------- Kernel A: attention per (i,h) --------------------------------------
// TDM stages K (256x64, strided) and V^T (64x256, contiguous) tiles into LDS;
// 8 waves x two 16-row j-tiles consume them with ds_load_b128 + WMMA.
__global__ __launch_bounds__(256)
void attn_kernel(const _Float16* __restrict__ q, const _Float16* __restrict__ kmat,
                 const _Float16* __restrict__ vt, const _Float16* __restrict__ g,
                 const float* __restrict__ mask, _Float16* __restrict__ og) {
    __shared__ __align__(16) _Float16 Plds[8][16][256];   // 64 KB (per-wave P staging)
    __shared__ __align__(16) _Float16 Klds[JJ * DD];      // 32 KB
    __shared__ __align__(16) _Float16 Vlds[DD * JJ];      // 32 KB
    int lane = threadIdx.x & 31;
    int wave = threadIdx.x >> 5;
    int i = blockIdx.x >> 3;
    int h = blockIdx.x & 7;
    int mrow = lane & 15;
    int kh   = (lane >> 4) * 8;
    int mhi  = (lane >> 4) << 3;

    const _Float16* qb  = q    + (size_t)i * JJ * HDIM + h * DD;
    const _Float16* kb_ = kmat + (size_t)i * JJ * HDIM + h * DD;
    const _Float16* vtb = vt   + (size_t)(i * HH + h) * DD * JJ;

    if (wave == 0) {
        // K tile: 256 rows of 64 halves, row stride 512 elements
        tdm_load_2d((unsigned)(unsigned long long)(uintptr_t)Klds, kb_,
                    /*dim0=*/DD, /*dim1=*/JJ, /*stride0=*/HDIM,
                    /*tile0=*/DD, /*tile1=*/JJ);
        // V^T tile: fully contiguous 64*256 halves
        tdm_load_2d((unsigned)(unsigned long long)(uintptr_t)Vlds, vtb,
                    /*dim0=*/DD * JJ, /*dim1=*/1, /*stride0=*/DD * JJ,
                    /*tile0=*/DD * JJ, /*tile1=*/1);
        __builtin_amdgcn_s_wait_tensorcnt(0);
    }
    __syncthreads();

    for (int jt = wave; jt < 16; jt += 8) {
        int j0 = jt * 16;
        // ---- S = q @ k^T : 16 tiles of 16x16, K=64 (2 chunks) ----
        v8f zero = {};
        v8f s[16];
#pragma unroll
        for (int t = 0; t < 16; t++) s[t] = zero;

        const _Float16* arow = qb + (size_t)(j0 + mrow) * HDIM + kh;
        v16h a0 = load_ab(arow);
        v16h a1 = load_ab(arow + 32);
#pragma unroll
        for (int t = 0; t < 16; t++) {
            const _Float16* brow = &Klds[(t * 16 + mrow) * DD + kh];
            v16h b0 = load_ab(brow);
            v16h b1 = load_ab(brow + 32);
            s[t] = wmma16(a0, b0, s[t]);
            s[t] = wmma16(a1, b1, s[t]);
            float biasv = 1e9f * (mask[(size_t)i * JJ + t * 16 + mrow] - 1.0f);
#pragma unroll
            for (int r = 0; r < 8; r++) s[t][r] += biasv;
        }

        // ---- softmax over k (each row's 16 cols live in one 16-lane half) ----
#pragma unroll
        for (int r = 0; r < 8; r++) {
            float m = s[0][r];
#pragma unroll
            for (int t = 1; t < 16; t++) m = fmaxf(m, s[t][r]);
#pragma unroll
            for (int off = 1; off < 16; off <<= 1) m = fmaxf(m, __shfl_xor(m, off, 32));
            float sum = 0.0f;
#pragma unroll
            for (int t = 0; t < 16; t++) { float e = __expf(s[t][r] - m); s[t][r] = e; sum += e; }
#pragma unroll
            for (int off = 1; off < 16; off <<= 1) sum += __shfl_xor(sum, off, 32);
            float inv = 1.0f / sum;
#pragma unroll
            for (int t = 0; t < 16; t++)
                Plds[wave][r + mhi][t * 16 + mrow] = (_Float16)(s[t][r] * inv);
        }
        asm volatile("s_wait_dscnt 0" ::: "memory");   // our P stores before our ds reads

        // ---- O = P @ v : 4 d-tiles, K=256 (8 chunks) ----
#pragma unroll
        for (int nt = 0; nt < 4; nt++) {
            v8f o = {};
            const _Float16* prow  = &Plds[wave][mrow][kh];
            const _Float16* brow0 = &Vlds[(nt * 16 + mrow) * JJ + kh];
#pragma unroll
            for (int kc = 0; kc < JJ; kc += 32) {
                v16h a = load_ab(prow + kc);
                v16h b = load_ab(brow0 + kc);
                o = wmma16(a, b, o);
            }
            int d = nt * 16 + mrow;
#pragma unroll
            for (int r = 0; r < 8; r++) {
                int j = j0 + r + mhi;
                size_t idx = ((size_t)(i * JJ + j)) * HDIM + h * DD + d;
                og[idx] = (_Float16)(o[r] * (float)g[idx]);
            }
        }
        asm volatile("s_wait_dscnt 0" ::: "memory");   // reads done before next jt overwrite
    }
}

// ---------- Kernel O: out = og[NROW,512] @ Wo + bo, wave does 32x32 -------------
__global__ __launch_bounds__(256)
void outproj_kernel(const _Float16* __restrict__ og, const _Float16* __restrict__ wot,
                    const float* __restrict__ bo, float* __restrict__ out) {
    int lane = threadIdx.x & 31;
    int wave = threadIdx.x >> 5;
    int tile = blockIdx.x * 8 + wave;       // 16384 wave tiles
    int tm = tile >> 3;                     // 2048 row tiles (32 rows)
    int tn = tile & 7;                      // 8 col tiles (32 cols)
    int row0 = tm * 32, col0 = tn * 32;
    int mrow = lane & 15;
    int kh   = (lane >> 4) * 8;
    int mhi  = (lane >> 4) << 3;

    const _Float16* a0p = og  + (size_t)(row0 + mrow) * HDIM + kh;
    const _Float16* a1p = a0p + (size_t)16 * HDIM;
    const _Float16* b0p = wot + (size_t)(col0 + mrow) * HDIM + kh;
    const _Float16* b1p = b0p + (size_t)16 * HDIM;

    v8f acc[2][2];
    acc[0][0] = {}; acc[0][1] = {}; acc[1][0] = {}; acc[1][1] = {};
#pragma unroll
    for (int kb = 0; kb < HDIM; kb += 32) {
        v16h a0 = load_ab(a0p + kb);
        v16h a1 = load_ab(a1p + kb);
        v16h b0 = load_ab(b0p + kb);
        v16h b1 = load_ab(b1p + kb);
        acc[0][0] = wmma16(a0, b0, acc[0][0]);
        acc[0][1] = wmma16(a0, b1, acc[0][1]);
        acc[1][0] = wmma16(a1, b0, acc[1][0]);
        acc[1][1] = wmma16(a1, b1, acc[1][1]);
    }
#pragma unroll
    for (int p = 0; p < 2; p++) {
#pragma unroll
        for (int qq = 0; qq < 2; qq++) {
            int colN = col0 + qq * 16 + mrow;
            float bias = bo[colN];
#pragma unroll
            for (int r = 0; r < 8; r++) {
                int row = row0 + p * 16 + r + mhi;
                out[(size_t)row * CDIM + colN] = acc[p][qq][r] + bias;
            }
        }
    }
}

// ---------- host ----------
extern "C" void kernel_launch(void* const* d_in, const int* in_sizes, int n_in,
                              void* d_out, int out_size, void* d_ws, size_t ws_size,
                              hipStream_t stream) {
    const float* x     = (const float*)d_in[0];
    const float* mask  = (const float*)d_in[1];
    const float* gamma = (const float*)d_in[2];
    const float* beta  = (const float*)d_in[3];
    const float* Wq    = (const float*)d_in[4];
    const float* Wk    = (const float*)d_in[5];
    const float* Wv    = (const float*)d_in[6];
    const float* Wg    = (const float*)d_in[7];
    const float* bg    = (const float*)d_in[8];
    const float* Wo    = (const float*)d_in[9];
    const float* bo    = (const float*)d_in[10];
    float* out = (float*)d_out;

    char* ws = (char*)d_ws;
    size_t off = 0;
    auto alloc = [&](size_t bytes) -> void* {
        void* p = ws + off;
        off += (bytes + 255) & ~(size_t)255;
        return p;
    };
    _Float16* xn   = (_Float16*)alloc((size_t)NROW * CDIM * 2);
    _Float16* WqT  = (_Float16*)alloc((size_t)HDIM * CDIM * 2);
    _Float16* WkT  = (_Float16*)alloc((size_t)HDIM * CDIM * 2);
    _Float16* WvT  = (_Float16*)alloc((size_t)HDIM * CDIM * 2);
    _Float16* WgT  = (_Float16*)alloc((size_t)HDIM * CDIM * 2);
    _Float16* WoT  = (_Float16*)alloc((size_t)CDIM * HDIM * 2);
    _Float16* qbuf = (_Float16*)alloc((size_t)NROW * HDIM * 2);
    _Float16* kbuf = (_Float16*)alloc((size_t)NROW * HDIM * 2);
    _Float16* vt   = (_Float16*)alloc((size_t)NROW * HDIM * 2);
    _Float16* gbuf = (_Float16*)alloc((size_t)NROW * HDIM * 2);
    _Float16* og   = (_Float16*)alloc((size_t)NROW * HDIM * 2);
    (void)ws_size; (void)in_sizes; (void)n_in; (void)out_size;

    // Weight transpose + f16 convert
    int wgrid = (CDIM * HDIM + 255) / 256;   // 512 blocks
    wtrans_kernel<<<wgrid, 256, 0, stream>>>(Wq, WqT, CDIM, HDIM);
    wtrans_kernel<<<wgrid, 256, 0, stream>>>(Wk, WkT, CDIM, HDIM);
    wtrans_kernel<<<wgrid, 256, 0, stream>>>(Wv, WvT, CDIM, HDIM);
    wtrans_kernel<<<wgrid, 256, 0, stream>>>(Wg, WgT, CDIM, HDIM);
    wtrans_kernel<<<wgrid, 256, 0, stream>>>(Wo, WoT, HDIM, CDIM);

    // LayerNorm
    ln_kernel<<<NROW, 256, 0, stream>>>(x, gamma, beta, xn);

    // Projections: 32768 wave-tiles / 8 waves = 4096 blocks each
    proj_kernel<<<4096, 256, 0, stream>>>(xn, WqT, bg, qbuf, vt, 0);
    proj_kernel<<<4096, 256, 0, stream>>>(xn, WkT, bg, kbuf, vt, 1);
    proj_kernel<<<4096, 256, 0, stream>>>(xn, WvT, bg, kbuf, vt, 2);
    proj_kernel<<<4096, 256, 0, stream>>>(xn, WgT, bg, gbuf, vt, 3);

    // Attention: one block per (i,h)
    attn_kernel<<<II * HH, 256, 0, stream>>>(qbuf, kbuf, vt, gbuf, mask, og);

    // Output projection: 16384 wave-tiles / 8 = 2048 blocks
    outproj_kernel<<<2048, 256, 0, stream>>>(og, WoT, bo, out);
}
